// MultiHeadSelfAttentionModule_22548578304189
// MI455X (gfx1250) — compile-verified, matrix operand order checked by reading, not run on hardware
//
#include <hip/hip_runtime.h>

#define S_LEN 1024
#define D_DIM 512
#define H_NUM 8
#define HD_DIM 64
#define B_NUM 8

typedef __bf16 bf16;
typedef __attribute__((ext_vector_type(8)))  __bf16 v8bf;
typedef __attribute__((ext_vector_type(16))) __bf16 v16bf;
typedef __attribute__((ext_vector_type(8)))  float  v8f;

#define WMMA_BF16(a, b, c) \
  __builtin_amdgcn_wmma_f32_16x16x32_bf16(false, (a), false, (b), (short)0, (c), false, false)

// Load a 16x32 bf16 A/B fragment slice for this lane.
// Caller passes p already offset by (lane<16 ? 0 : 8) along K.
// elements 0..7  <- K kbase+0..7   (VGPRs 0..3)
// elements 8..15 <- K kbase+16..23 (VGPRs 4..7)
static __device__ __forceinline__ v16bf frag_ld(const bf16* __restrict__ p) {
  v8bf lo = *reinterpret_cast<const v8bf*>(p);
  v8bf hi = *reinterpret_cast<const v8bf*>(p + 16);
  v16bf r;
#pragma unroll
  for (int i = 0; i < 8; ++i) { r[i] = lo[i]; r[i + 8] = hi[i]; }
  return r;
}

// ---------------- LayerNorm: f32 in -> bf16 out ----------------
__global__ __launch_bounds__(128) void ln_kernel(const float* __restrict__ x,
                                                 const float* __restrict__ g,
                                                 const float* __restrict__ bta,
                                                 bf16* __restrict__ xn) {
  const int row = blockIdx.x;          // B*S rows
  const int tid = threadIdx.x;         // 128 threads, 4 elems each
  __shared__ float red[128];
  const float* xr = x + (size_t)row * D_DIM;
  float v[4];
  float s = 0.f;
#pragma unroll
  for (int i = 0; i < 4; ++i) { v[i] = xr[tid * 4 + i]; s += v[i]; }
  red[tid] = s; __syncthreads();
  for (int off = 64; off > 0; off >>= 1) { if (tid < off) red[tid] += red[tid + off]; __syncthreads(); }
  float mu = red[0] * (1.0f / D_DIM);
  __syncthreads();
  float s2 = 0.f;
#pragma unroll
  for (int i = 0; i < 4; ++i) { float d = v[i] - mu; s2 += d * d; }
  red[tid] = s2; __syncthreads();
  for (int off = 64; off > 0; off >>= 1) { if (tid < off) red[tid] += red[tid + off]; __syncthreads(); }
  float rstd = rsqrtf(red[0] * (1.0f / D_DIM) + 1e-5f);
#pragma unroll
  for (int i = 0; i < 4; ++i) {
    int c = tid * 4 + i;
    xn[(size_t)row * D_DIM + c] = (bf16)((v[i] - mu) * rstd * g[c] + bta[c]);
  }
}

// ------------- Weight convert + transpose: Wt[n][k] = (bf16)W[k][n] -------------
__global__ __launch_bounds__(256) void wconv_kernel(const float* __restrict__ W,
                                                    bf16* __restrict__ Wt) {
  int e = blockIdx.x * blockDim.x + threadIdx.x;
  if (e >= D_DIM * D_DIM) return;
  int n = e / D_DIM, k = e % D_DIM;
  Wt[e] = (bf16)W[(size_t)k * D_DIM + n];
}

// ------------- Sinusoidal PE directly in bf16 -------------
__global__ __launch_bounds__(256) void pe_kernel(bf16* __restrict__ pe) {
  int e = blockIdx.x * blockDim.x + threadIdx.x;   // S * D/2 pairs
  if (e >= S_LEN * (D_DIM / 2)) return;
  int s = e / (D_DIM / 2), j = e % (D_DIM / 2);
  float div = __expf((float)(2 * j) * (-9.210340371976184f / (float)D_DIM));
  float ang = (float)s * div;
  pe[(size_t)s * D_DIM + 2 * j]     = (bf16)__sinf(ang);
  pe[(size_t)s * D_DIM + 2 * j + 1] = (bf16)__cosf(ang);
}

// ------------- V transpose: vt[b][h][hd][s] = v[b][s][h*64+hd] -------------
__global__ __launch_bounds__(256) void vtrans_kernel(const bf16* __restrict__ v,
                                                     bf16* __restrict__ vt) {
  int e = blockIdx.x * blockDim.x + threadIdx.x;   // B*S*D
  if (e >= B_NUM * S_LEN * D_DIM) return;
  int d = e % D_DIM;
  int s = (e / D_DIM) % S_LEN;
  int b = e / (D_DIM * S_LEN);
  int h = d >> 6, hd = d & 63;
  vt[(((size_t)b * H_NUM + h) * HD_DIM + hd) * S_LEN + s] = v[e];
}

// ------------- WMMA GEMM: C(MxN) = A(MxK) @ Bt^T + bias -------------
// Bt stored N-major (N x K, contiguous K). One wave computes a 32x64 tile,
// software-pipelined: next k-step fragments are prefetched while current
// WMMAs execute, so loads overlap the matrix pipe.
// mode 0: out1 = bf16(acc+bias)
// mode 1: out1 = bf16(acc+bias+add1[n]); out2 = bf16(acc+bias+add2[n])   (Q path)
// mode 2: outF = f32(acc+bias)                                            (final O)
__global__ __launch_bounds__(128) void wmma_gemm_kernel(
    const bf16* __restrict__ A, const bf16* __restrict__ Bt,
    const float* __restrict__ bias,
    bf16* __restrict__ out1, bf16* __restrict__ out2,
    const float* __restrict__ add1, const float* __restrict__ add2,
    float* __restrict__ outF, int M, int N, int K, int mode) {
  const int wid  = blockIdx.x * (blockDim.x >> 5) + (threadIdx.x >> 5);
  const int lane = threadIdx.x & 31;
  const int nChunks = N >> 6;
  const int mTiles  = M >> 5;            // 32-row tiles
  if (wid >= mTiles * nChunks) return;
  const int mt = wid / nChunks, nc = wid % nChunks;
  const int m0 = mt << 5, n0 = nc << 6;
  const int colL  = lane & 15;
  const int koff  = (lane < 16) ? 0 : 8;
  const int rbase = (lane < 16) ? 0 : 8;

  v8f acc[2][4];
#pragma unroll
  for (int t = 0; t < 2; ++t)
#pragma unroll
    for (int j = 0; j < 4; ++j)
#pragma unroll
      for (int r = 0; r < 8; ++r) acc[t][j][r] = 0.f;

  const bf16* aR0 = A + (size_t)(m0 + colL) * K + koff;
  const bf16* aR1 = aR0 + (size_t)16 * K;
  const bf16* bR  = Bt + (size_t)(n0 + colL) * K + koff;

  // prologue: first k-step fragments
  v16bf a0 = frag_ld(aR0), a1 = frag_ld(aR1);
  v16bf b0 = frag_ld(bR);
  v16bf b1 = frag_ld(bR + (size_t)16 * K);
  v16bf b2 = frag_ld(bR + (size_t)32 * K);
  v16bf b3 = frag_ld(bR + (size_t)48 * K);

  for (int k0 = 32; k0 <= K; k0 += 32) {
    // prefetch next k-step while current WMMAs run
    v16bf a0n = a0, a1n = a1, b0n = b0, b1n = b1, b2n = b2, b3n = b3;
    if (k0 < K) {
      a0n = frag_ld(aR0 + k0);
      a1n = frag_ld(aR1 + k0);
      b0n = frag_ld(bR + k0);
      b1n = frag_ld(bR + (size_t)16 * K + k0);
      b2n = frag_ld(bR + (size_t)32 * K + k0);
      b3n = frag_ld(bR + (size_t)48 * K + k0);
    }
    acc[0][0] = WMMA_BF16(a0, b0, acc[0][0]);
    acc[1][0] = WMMA_BF16(a1, b0, acc[1][0]);
    acc[0][1] = WMMA_BF16(a0, b1, acc[0][1]);
    acc[1][1] = WMMA_BF16(a1, b1, acc[1][1]);
    acc[0][2] = WMMA_BF16(a0, b2, acc[0][2]);
    acc[1][2] = WMMA_BF16(a1, b2, acc[1][2]);
    acc[0][3] = WMMA_BF16(a0, b3, acc[0][3]);
    acc[1][3] = WMMA_BF16(a1, b3, acc[1][3]);
    a0 = a0n; a1 = a1n; b0 = b0n; b1 = b1n; b2 = b2n; b3 = b3n;
  }

#pragma unroll
  for (int t = 0; t < 2; ++t) {
#pragma unroll
    for (int j = 0; j < 4; ++j) {
      int n = n0 + 16 * j + colL;
      float bb = bias ? bias[n] : 0.0f;
      float a1e = (mode == 1) ? add1[n] : 0.0f;
      float a2e = (mode == 1) ? add2[n] : 0.0f;
#pragma unroll
      for (int r = 0; r < 8; ++r) {
        int m = m0 + 16 * t + rbase + r;
        float val = acc[t][j][r] + bb;
        size_t idx = (size_t)m * N + n;
        if (mode == 2)      outF[idx] = val;
        else if (mode == 1) { out1[idx] = (bf16)(val + a1e); out2[idx] = (bf16)(val + a2e); }
        else                out1[idx] = (bf16)val;
      }
    }
  }
}

// ------------- Fused relative-position attention -------------
// One workgroup (4 waves) per (b, h, 16-row query tile).
// LDS: ps strip (16x1024 f32) + extra ps row (1024 f32) + score strip (16x1024 f32)
//      + 256 f32 scratch. bf16 attn strip aliases the (dead after Phase B) ps strip.
__global__ __launch_bounds__(128) void attn_kernel(
    const bf16* __restrict__ qu, const bf16* __restrict__ qv,
    const bf16* __restrict__ kmat, const bf16* __restrict__ pmat,
    const bf16* __restrict__ vt, bf16* __restrict__ ctx) {
  extern __shared__ char smem_raw[];
  float* psBuf   = (float*)smem_raw;        // 16*1024
  float* psExtra = psBuf + 16 * S_LEN;      // 1024
  float* scores  = psExtra + S_LEN;         // 16*1024
  float* red     = scores + 16 * S_LEN;     // 256
  bf16*  attnBf  = (bf16*)smem_raw;         // 16*1024 bf16, aliases psBuf (dead then)

  const int tid  = threadIdx.x;
  const int lane = tid & 31;
  const int wave = tid >> 5;
  const int qt = blockIdx.x % (S_LEN / 16);
  const int bh = blockIdx.x / (S_LEN / 16);
  const int b  = bh >> 3, h = bh & 7;
  const int q0 = qt * 16;
  const int colL  = lane & 15;
  const int koff  = (lane < 16) ? 0 : 8;
  const int rbase = (lane < 16) ? 0 : 8;
  const size_t headOff = (size_t)h * HD_DIM;

  // Preload A fragments (rows q0..q0+15, K = head dim 64 -> 2 steps of 32).
  const bf16* qvRow = qv + ((size_t)(b * S_LEN) + q0 + colL) * D_DIM + headOff + koff;
  const bf16* quRow = qu + ((size_t)(b * S_LEN) + q0 + colL) * D_DIM + headOff + koff;
  v16bf aqv0 = frag_ld(qvRow), aqv1 = frag_ld(qvRow + 32);
  v16bf aqu0 = frag_ld(quRow), aqu1 = frag_ld(quRow + 32);

  // Stage (q+v) row q0+16 (needed by the exact relative shift) into red[0..63].
  if (q0 + 16 < S_LEN && tid < HD_DIM)
    red[tid] = (float)qv[((size_t)(b * S_LEN) + q0 + 16) * D_DIM + headOff + tid];

  // Phase A: ps rows q0..q0+15 over all S columns, via WMMA (pipelined B frags).
  {
    int ct = wave;
    const bf16* pb = pmat + (size_t)(ct * 16 + colL) * D_DIM + headOff + koff;
    v16bf b0 = frag_ld(pb), b1 = frag_ld(pb + 32);
    while (ct < S_LEN / 16) {
      int ctn = ct + 4;
      v16bf b0n = b0, b1n = b1;
      if (ctn < S_LEN / 16) {
        const bf16* pn = pmat + (size_t)(ctn * 16 + colL) * D_DIM + headOff + koff;
        b0n = frag_ld(pn); b1n = frag_ld(pn + 32);
      }
      v8f c;
#pragma unroll
      for (int r = 0; r < 8; ++r) c[r] = 0.f;
      c = WMMA_BF16(aqv0, b0, c);
      c = WMMA_BF16(aqv1, b1, c);
      int n0 = ct * 16;
#pragma unroll
      for (int r = 0; r < 8; ++r) psBuf[(rbase + r) * S_LEN + n0 + colL] = c[r];
      b0 = b0n; b1 = b1n; ct = ctn;
    }
  }
  __syncthreads();

  // Phase A2: ps row q0+16 via VALU dot products (1024 x 64 MACs, vectorized).
  if (q0 + 16 < S_LEN) {
    for (int i = 0; i < 8; ++i) {
      int c = tid * 8 + i;
      const bf16* pr = pmat + (size_t)c * D_DIM + headOff;
      float dot = 0.f;
#pragma unroll
      for (int d0 = 0; d0 < HD_DIM; d0 += 8) {
        v8bf pv8 = *reinterpret_cast<const v8bf*>(pr + d0);
#pragma unroll
        for (int e = 0; e < 8; ++e) dot += red[d0 + e] * (float)pv8[e];
      }
      psExtra[c] = dot;
    }
  }
  __syncthreads();

  // Phase B: content scores via WMMA (pipelined) + exact relative shift + scale.
  {
    int ct = wave;
    const bf16* kb0 = kmat + ((size_t)(b * S_LEN) + ct * 16 + colL) * D_DIM + headOff + koff;
    v16bf b0 = frag_ld(kb0), b1 = frag_ld(kb0 + 32);
    while (ct < S_LEN / 16) {
      int ctn = ct + 4;
      v16bf b0n = b0, b1n = b1;
      if (ctn < S_LEN / 16) {
        const bf16* kn = kmat + ((size_t)(b * S_LEN) + ctn * 16 + colL) * D_DIM + headOff + koff;
        b0n = frag_ld(kn); b1n = frag_ld(kn + 32);
      }
      v8f c;
#pragma unroll
      for (int r = 0; r < 8; ++r) c[r] = 0.f;
      c = WMMA_BF16(aqu0, b0, c);
      c = WMMA_BF16(aqu1, b1, c);
      int cg = ct * 16 + colL;
#pragma unroll
      for (int r = 0; r < 8; ++r) {
        int q = q0 + rbase + r;
        float pv;
        if (cg <= q) {
          pv = psBuf[(q - q0) * S_LEN + (S_LEN - 1 - q + cg)];
        } else if (cg == q + 1) {
          pv = 0.0f;
        } else {
          int rl = q - q0 + 1;
          int sc = cg - q - 2;
          pv = (rl < 16) ? psBuf[rl * S_LEN + sc] : psExtra[sc];
        }
        scores[(q - q0) * S_LEN + cg] = (c[r] + pv) * 0.125f;   // 1/sqrt(64)
      }
      b0 = b0n; b1 = b1n; ct = ctn;
    }
  }
  __syncthreads();

  // Row-wise softmax over S in LDS (float4 passes). 8 threads per row, 128 cols each.
  // Normalized probabilities are written as bf16 into attnBf (aliases dead psBuf).
  {
    int rowL = tid >> 3, part = tid & 7;
    float4* sr4 = (float4*)(scores + rowL * S_LEN + part * 128);
    float m = -3.4e38f;
    for (int i = 0; i < 32; ++i) {
      float4 v = sr4[i];
      m = fmaxf(m, fmaxf(fmaxf(v.x, v.y), fmaxf(v.z, v.w)));
    }
    red[rowL * 8 + part] = m;
    __syncthreads();
    float rm = red[rowL * 8];
#pragma unroll
    for (int i = 1; i < 8; ++i) rm = fmaxf(rm, red[rowL * 8 + i]);
    float sum = 0.f;
    for (int i = 0; i < 32; ++i) {
      float4 v = sr4[i];
      v.x = __expf(v.x - rm); v.y = __expf(v.y - rm);
      v.z = __expf(v.z - rm); v.w = __expf(v.w - rm);
      sum += (v.x + v.y) + (v.z + v.w);
      sr4[i] = v;
    }
    red[128 + rowL * 8 + part] = sum;
    __syncthreads();
    float rs = 0.f;
#pragma unroll
    for (int i = 0; i < 8; ++i) rs += red[128 + rowL * 8 + i];
    float inv = 1.0f / rs;
    bf16* ab = attnBf + rowL * S_LEN + part * 128;
    for (int i = 0; i < 32; ++i) {
      float4 v = sr4[i];
      ab[i * 4 + 0] = (bf16)(v.x * inv);
      ab[i * 4 + 1] = (bf16)(v.y * inv);
      ab[i * 4 + 2] = (bf16)(v.z * inv);
      ab[i * 4 + 3] = (bf16)(v.w * inv);
    }
  }
  __syncthreads();

  // Phase C: ctx = attn @ V. Each wave owns one 16-wide head-dim subtile.
  // A-fragments come straight from bf16 LDS (ds_load_b128, no conversions);
  // V fragments are pipelined from global.
  {
    int n0h = wave * 16;
    v8f acc;
#pragma unroll
    for (int r = 0; r < 8; ++r) acc[r] = 0.f;
    const bf16* vrow = vt + ((size_t)bh * HD_DIM + n0h + colL) * S_LEN + koff;
    const bf16* arow = attnBf + colL * S_LEN + koff;
    v16bf bfr = frag_ld(vrow);
    for (int k0 = 0; k0 < S_LEN; k0 += 32) {
      v16bf bn = bfr;
      if (k0 + 32 < S_LEN) bn = frag_ld(vrow + k0 + 32);
      v16bf a = frag_ld(arow + k0);
      acc = WMMA_BF16(a, bfr, acc);
      bfr = bn;
    }
#pragma unroll
    for (int r = 0; r < 8; ++r) {
      int q = q0 + rbase + r;
      ctx[((size_t)(b * S_LEN) + q) * D_DIM + headOff + n0h + colL] = (bf16)acc[r];
    }
  }
}

// ---------------------------------------------------------------------------
extern "C" void kernel_launch(void* const* d_in, const int* in_sizes, int n_in,
                              void* d_out, int out_size, void* d_ws, size_t ws_size,
                              hipStream_t stream) {
  const float* x   = (const float*)d_in[0];
  const float* Wq  = (const float*)d_in[1];
  const float* bq  = (const float*)d_in[2];
  const float* Wk  = (const float*)d_in[3];
  const float* bk  = (const float*)d_in[4];
  const float* Wv  = (const float*)d_in[5];
  const float* bv  = (const float*)d_in[6];
  const float* Wo  = (const float*)d_in[7];
  const float* bo  = (const float*)d_in[8];
  const float* Wp  = (const float*)d_in[9];
  const float* ub  = (const float*)d_in[10];
  const float* vb  = (const float*)d_in[11];
  const float* lng = (const float*)d_in[12];
  const float* lnb = (const float*)d_in[13];
  float* out = (float*)d_out;
  (void)in_sizes; (void)n_in; (void)out_size; (void)ws_size;

  char* ws = (char*)d_ws;
  size_t off = 0;
  auto alloc = [&](size_t bytes) -> char* {
    char* p = ws + off;
    off += (bytes + 255) & ~(size_t)255;
    return p;
  };
  const size_t actB = (size_t)B_NUM * S_LEN * D_DIM * sizeof(bf16);   // 8 MB
  const size_t wB   = (size_t)D_DIM * D_DIM * sizeof(bf16);           // 512 KB

  bf16* xn   = (bf16*)alloc(actB);
  bf16* WtQ  = (bf16*)alloc(wB);
  bf16* WtK  = (bf16*)alloc(wB);
  bf16* WtV  = (bf16*)alloc(wB);
  bf16* WtO  = (bf16*)alloc(wB);
  bf16* WtP  = (bf16*)alloc(wB);
  bf16* peb  = (bf16*)alloc((size_t)S_LEN * D_DIM * sizeof(bf16));
  bf16* quB  = (bf16*)alloc(actB);
  bf16* qvB  = (bf16*)alloc(actB);
  bf16* kB   = (bf16*)alloc(actB);
  bf16* vB   = (bf16*)alloc(actB);
  bf16* vtB  = (bf16*)alloc(actB);
  bf16* pB   = (bf16*)alloc((size_t)S_LEN * D_DIM * sizeof(bf16));
  bf16* ctxB = (bf16*)alloc(actB);

  // 1) LayerNorm -> bf16 xn
  ln_kernel<<<B_NUM * S_LEN, 128, 0, stream>>>(x, lng, lnb, xn);

  // 2) Weight convert+transpose (f32 -> bf16, N-major)
  int wBlocks = (D_DIM * D_DIM + 255) / 256;
  wconv_kernel<<<wBlocks, 256, 0, stream>>>(Wq, WtQ);
  wconv_kernel<<<wBlocks, 256, 0, stream>>>(Wk, WtK);
  wconv_kernel<<<wBlocks, 256, 0, stream>>>(Wv, WtV);
  wconv_kernel<<<wBlocks, 256, 0, stream>>>(Wo, WtO);
  wconv_kernel<<<wBlocks, 256, 0, stream>>>(Wp, WtP);

  // 3) Sinusoidal positional encoding
  pe_kernel<<<(S_LEN * (D_DIM / 2) + 255) / 256, 256, 0, stream>>>(peb);

  // 4) Projections (all K=N=512); one wave = 32x64 tile
  auto launch_gemm = [&](const bf16* A, const bf16* Bt, const float* bias,
                         bf16* o1, bf16* o2, const float* a1, const float* a2,
                         float* oF, int M, int mode) {
    int waves = (M / 32) * (D_DIM / 64);
    wmma_gemm_kernel<<<(waves + 3) / 4, 128, 0, stream>>>(
        A, Bt, bias, o1, o2, a1, a2, oF, M, D_DIM, D_DIM, mode);
  };
  launch_gemm(xn, WtQ, bq, quB, qvB, ub, vb, nullptr, B_NUM * S_LEN, 1); // q+u, q+v
  launch_gemm(xn, WtK, bk, kB, nullptr, nullptr, nullptr, nullptr, B_NUM * S_LEN, 0);
  launch_gemm(xn, WtV, bv, vB, nullptr, nullptr, nullptr, nullptr, B_NUM * S_LEN, 0);
  launch_gemm(peb, WtP, nullptr, pB, nullptr, nullptr, nullptr, nullptr, S_LEN, 0);

  // 5) Transpose V for WMMA B-fragment contiguity
  vtrans_kernel<<<(B_NUM * S_LEN * D_DIM + 255) / 256, 256, 0, stream>>>(vB, vtB);

  // 6) Fused attention (scores + exact rel-shift + softmax + attn@V, all in LDS)
  size_t shBytes = (size_t)(16 * S_LEN + S_LEN + 16 * S_LEN + 256) * sizeof(float); // ~136 KB of 320 KB WGP LDS
  attn_kernel<<<B_NUM * H_NUM * (S_LEN / 16), 128, shBytes, stream>>>(
      quB, qvB, kB, pB, vtB, ctxB);

  // 7) Output projection -> f32 d_out
  launch_gemm(ctxB, WtO, bo, nullptr, nullptr, nullptr, nullptr, out, B_NUM * S_LEN, 2);
}
